// BitNetLinear_89489938580085
// MI455X (gfx1250) — matrix operand order repaired and checked by reading, not compiled, not run on hardware
//
#include <hip/hip_runtime.h>

typedef _Float16     v16h  __attribute__((ext_vector_type(16)));
typedef _Float16     v8h   __attribute__((ext_vector_type(8)));
typedef _Float16     v4h   __attribute__((ext_vector_type(4)));
typedef float        v8f   __attribute__((ext_vector_type(8)));
typedef float        v4f   __attribute__((ext_vector_type(4)));
typedef unsigned int v8u   __attribute__((ext_vector_type(8)));
typedef unsigned int v4u   __attribute__((ext_vector_type(4)));
typedef unsigned int v2u   __attribute__((ext_vector_type(2)));
typedef int          i32x4 __attribute__((vector_size(16)));   // builtin param type

#define M_DIM 8192
#define N_DIM 16384
#define K_DIM 4096

#if __has_builtin(__builtin_amdgcn_global_load_async_to_lds_b128)
#define HAVE_ASYNC 1
#endif

// ---- helpers ---------------------------------------------------------------
__device__ __forceinline__ void cp16_async(const void* g, void* l) {
#ifdef HAVE_ASYNC
    __builtin_amdgcn_global_load_async_to_lds_b128(
        (__attribute__((address_space(1))) i32x4*)(unsigned long long)g,
        (__attribute__((address_space(3))) i32x4*)(unsigned int)(unsigned long long)l,
        0, 0);
#else
    *(v4f*)l = *(const v4f*)g;                // fallback: load + ds_store
#endif
}

template <int N>
__device__ __forceinline__ void wait_async() {
#ifdef HAVE_ASYNC
#if __has_builtin(__builtin_amdgcn_s_wait_asynccnt)
    __builtin_amdgcn_s_wait_asynccnt((unsigned short)N);
#else
    asm volatile("s_wait_asynccnt %0" :: "i"(N) : "memory");
#endif
#endif
}

// sign(a),sign(b) as packed f16 {±1,±1}: 0x3C00 | signbit (w==0 -> +1, measure-zero)
__device__ __forceinline__ unsigned int sgn_pk(unsigned int a, unsigned int b) {
    return 0x3C003C00u | ((a >> 16) & 0x8000u) | (b & 0x80000000u);
}

// ---------------------------------------------------------------------------
// Pass 0: zero alpha accumulator
// ---------------------------------------------------------------------------
__global__ void bitnet_zero_ws(float* ws) {
    if (threadIdx.x == 0) ws[0] = 0.0f;
}

// ---------------------------------------------------------------------------
// Pass 1a: x f32 -> f16, streaming both ways (done ONCE, not per block)
// ---------------------------------------------------------------------------
__global__ __launch_bounds__(256) void bitnet_quant_x(const float* __restrict__ X,
                                                      _Float16* __restrict__ Xh,
                                                      unsigned int nQuads) {
    unsigned int gidx   = blockIdx.x * 256u + threadIdx.x;
    unsigned int stride = gridDim.x * 256u;
    const v4f* X4 = (const v4f*)X;
    v4h*       H4 = (v4h*)Xh;
    for (unsigned int i = gidx; i < nQuads; i += stride) {
        v4f v = __builtin_nontemporal_load(X4 + i);
        v4h h = { (_Float16)v.x, (_Float16)v.y, (_Float16)v.z, (_Float16)v.w };
        __builtin_nontemporal_store(h, H4 + i);
    }
}

// ---------------------------------------------------------------------------
// Pass 1b: W f32 -> sign(W) f16 (+-1), fused alpha = sum(|W|)  (done ONCE)
// ---------------------------------------------------------------------------
__global__ __launch_bounds__(256) void bitnet_quant_w(const float* __restrict__ W,
                                                      _Float16* __restrict__ Wh,
                                                      float* __restrict__ ws,
                                                      unsigned int nQuads) {
    __shared__ float red[256];
    unsigned int tid    = threadIdx.x;
    unsigned int gidx   = blockIdx.x * 256u + tid;
    unsigned int stride = gridDim.x * 256u;

    float s = 0.0f;
    const v4u* W4 = (const v4u*)W;
    v2u*       S2 = (v2u*)Wh;
    for (unsigned int i = gidx; i < nQuads; i += stride) {
        v4u r = __builtin_nontemporal_load(W4 + i);
        v2u pk = { sgn_pk(r.x, r.y), sgn_pk(r.z, r.w) };
        __builtin_nontemporal_store(pk, S2 + i);
        s += __builtin_bit_cast(float, r.x & 0x7fffffffu) +
             __builtin_bit_cast(float, r.y & 0x7fffffffu) +
             __builtin_bit_cast(float, r.z & 0x7fffffffu) +
             __builtin_bit_cast(float, r.w & 0x7fffffffu);
    }
    red[tid] = s;
    __syncthreads();
#pragma unroll
    for (int off = 128; off > 0; off >>= 1) {
        if (tid < (unsigned)off) red[tid] += red[tid + off];
        __syncthreads();
    }
    if (tid == 0) atomicAdd(ws, red[0]);
}

// ---------------------------------------------------------------------------
// Pass 2 (primary): f16-operand WMMA GEMM, zero conversion VALU in the loop.
// Block tile 128(M) x 256(N), BK=64, 8 wave32 in 2x4; wave tile 64x64 = 4x4 wmma
// (16 accumulators = 128 VGPRs -> LDS fragment traffic 0.0625 B/MAC)
// ---------------------------------------------------------------------------
#define PBM 128
#define PBN 256
#define PBK 64
#define ASTR 72   // f16 staging stride: 144B rows, 16B-aligned chunks, gcd(36,64)=4

__global__ __launch_bounds__(256) void bitnet_gemm16(const _Float16* __restrict__ Xh,
                                                     const _Float16* __restrict__ Wh,
                                                     const float* __restrict__ bias,
                                                     const float* __restrict__ wsAlpha,
                                                     float* __restrict__ Out) {
    __shared__ _Float16 Ah[2][PBM * ASTR];   // 36 KB
    __shared__ _Float16 Bh[2][PBN * ASTR];   // 72 KB

    const int tid  = threadIdx.x;
    const int lane = tid & 31;
    const int wave = tid >> 5;
    const int wm   = wave & 1;               // 0..1 -> 64-row slice
    const int wn   = wave >> 1;              // 0..3 -> 64-col slice

    const int mBlock = blockIdx.x * PBM;     // M fastest: W panel L2-resident
    const int nBlock = blockIdx.y * PBN;

    const int halfSel = lane >> 4;
    const int l16     = lane & 15;

    const int loadRow  = tid >> 3;           // 0..31
    const int loadQ    = (tid & 7) * 8;      // f16 offset of 16B chunk in 64-wide row

    v8f acc[4][4];
#pragma unroll
    for (int t = 0; t < 4; ++t)
#pragma unroll
        for (int u = 0; u < 4; ++u)
            acc[t][u] = (v8f){0.f,0.f,0.f,0.f,0.f,0.f,0.f,0.f};

    // 12 async b128 per thread per K-tile (4 for A, 8 for B)
    auto issue_tile = [&](int kb, int buf) {
#pragma unroll
        for (int r = 0; r < 4; ++r) {
            const int row = loadRow + r * 32;
            cp16_async(Xh + (size_t)(mBlock + row) * K_DIM + kb + loadQ,
                       &Ah[buf][row * ASTR + loadQ]);
        }
#pragma unroll
        for (int r = 0; r < 8; ++r) {
            const int row = loadRow + r * 32;
            cp16_async(Wh + (size_t)(nBlock + row) * K_DIM + kb + loadQ,
                       &Bh[buf][row * ASTR + loadQ]);
        }
    };

    issue_tile(0, 0);

    int buf = 0;
    for (int kb = 0; kb < K_DIM; kb += PBK, buf ^= 1) {
        const bool more = (kb + PBK) < K_DIM;
        if (more) {
            issue_tile(kb + PBK, buf ^ 1);
            wait_async<12>();                // current tile's copies complete
        } else {
            wait_async<0>();
        }
        __syncthreads();

#pragma unroll
        for (int s = 0; s < 2; ++s) {        // two 32-deep wmma steps per K-tile
            // A 16x32 f16: lanes 0-15 K 0-7 & 16-23; lanes 16-31 K 8-15 & 24-31
            v16h af[4];
            const int ak0 = s * 32 + halfSel * 8;
#pragma unroll
            for (int t = 0; t < 4; ++t) {
                const _Float16* p = &Ah[buf][(wm * 64 + t * 16 + l16) * ASTR + ak0];
                v8h lo = *(const v8h*)(p);
                v8h hi = *(const v8h*)(p + 16);
                af[t] = __builtin_shufflevector(lo, hi,
                            0,1,2,3,4,5,6,7,8,9,10,11,12,13,14,15);
            }
            // B 32x16 f16: lane n = column; lanes 0-15 K 0-7 & 16-23; lanes 16-31 K 8-15 & 24-31
            v16h bf[4];
            const int bk0 = s * 32 + halfSel * 8;
#pragma unroll
            for (int u = 0; u < 4; ++u) {
                const _Float16* p = &Bh[buf][(wn * 64 + u * 16 + l16) * ASTR + bk0];
                v8h lo = *(const v8h*)(p);
                v8h hi = *(const v8h*)(p + 16);
                bf[u] = __builtin_shufflevector(lo, hi,
                            0,1,2,3,4,5,6,7,8,9,10,11,12,13,14,15);
            }
#pragma unroll
            for (int t = 0; t < 4; ++t)
#pragma unroll
                for (int u = 0; u < 4; ++u)
                    acc[t][u] = __builtin_amdgcn_wmma_f32_16x16x32_f16(
                        false, af[t], false, bf[u],
                        (short)0, acc[t][u], false, false);
        }
        __syncthreads();
    }

    const float alpha = wsAlpha[0] * (1.0f / ((float)N_DIM * (float)K_DIM));
#pragma unroll
    for (int u = 0; u < 4; ++u) {
        const int col = nBlock + wn * 64 + u * 16 + l16;
        const float bv = bias[col];
#pragma unroll
        for (int t = 0; t < 4; ++t) {
            const int rowBase = mBlock + wm * 64 + t * 16 + halfSel * 8;
#pragma unroll
            for (int j = 0; j < 8; ++j)
                __builtin_nontemporal_store((acc[t][u][j] + bv) * alpha,
                    &Out[(size_t)(rowBase + j) * N_DIM + col]);
        }
    }
}

// ---------------------------------------------------------------------------
// Fallback pass 1: alpha only (when ws too small for quantized operands)
// ---------------------------------------------------------------------------
__global__ __launch_bounds__(256) void bitnet_absum(const float* __restrict__ W,
                                                    float* __restrict__ ws,
                                                    unsigned int nQuads) {
    __shared__ float red[256];
    unsigned int tid    = threadIdx.x;
    unsigned int gidx   = blockIdx.x * 256u + tid;
    unsigned int stride = gridDim.x * 256u;
    float s = 0.0f;
    const v4f* W4 = (const v4f*)W;
    for (unsigned int i = gidx; i < nQuads; i += stride) {
        v4f v = __builtin_nontemporal_load(W4 + i);
        s += __builtin_fabsf(v.x) + __builtin_fabsf(v.y) +
             __builtin_fabsf(v.z) + __builtin_fabsf(v.w);
    }
    red[tid] = s;
    __syncthreads();
#pragma unroll
    for (int off = 128; off > 0; off >>= 1) {
        if (tid < (unsigned)off) red[tid] += red[tid + off];
        __syncthreads();
    }
    if (tid == 0) atomicAdd(ws, red[0]);
}

// ---------------------------------------------------------------------------
// Fallback pass 2: inline-convert GEMM (round-3 kernel), f32 staging
// ---------------------------------------------------------------------------
#define FBM 128
#define FBN 128
#define FBK 32
#define SSTR 36

__global__ __launch_bounds__(256) void bitnet_gemm_fb(const float* __restrict__ X,
                                                      const float* __restrict__ W,
                                                      const float* __restrict__ bias,
                                                      const float* __restrict__ wsAlpha,
                                                      float* __restrict__ Out) {
    __shared__ float Xs[2][FBM * SSTR];
    __shared__ float Ws[2][FBN * SSTR];

    const int tid  = threadIdx.x;
    const int lane = tid & 31;
    const int wave = tid >> 5;
    const int wm   = wave & 1;
    const int wn   = wave >> 1;
    const int mBlock = blockIdx.x * FBM;
    const int nBlock = blockIdx.y * FBN;
    const int halfSel = lane >> 4;
    const int l16     = lane & 15;
    const int loadRow  = tid >> 3;
    const int loadQuad = tid & 7;

    v8f acc[4][2];
#pragma unroll
    for (int t = 0; t < 4; ++t)
#pragma unroll
        for (int u = 0; u < 2; ++u)
            acc[t][u] = (v8f){0.f,0.f,0.f,0.f,0.f,0.f,0.f,0.f};

    auto issue_tile = [&](int kb, int buf) {
#pragma unroll
        for (int r = 0; r < 4; ++r) {
            const int row = loadRow + r * 32;
            cp16_async(X + (size_t)(mBlock + row) * K_DIM + kb + loadQuad * 4,
                       &Xs[buf][row * SSTR + loadQuad * 4]);
            cp16_async(W + (size_t)(nBlock + row) * K_DIM + kb + loadQuad * 4,
                       &Ws[buf][row * SSTR + loadQuad * 4]);
        }
    };

    issue_tile(0, 0);
    int buf = 0;
    for (int kb = 0; kb < K_DIM; kb += FBK, buf ^= 1) {
        const bool more = (kb + FBK) < K_DIM;
        if (more) { issue_tile(kb + FBK, buf ^ 1); wait_async<8>(); }
        else      { wait_async<0>(); }
        __syncthreads();

        const float* As = Xs[buf];
        const float* Bs = Ws[buf];
        v16h afrag[4];
        const int ak0 = halfSel * 8;
#pragma unroll
        for (int t = 0; t < 4; ++t) {
            const float* p = &As[(wm * 64 + t * 16 + l16) * SSTR + ak0];
            v4f a0 = *(const v4f*)(p);
            v4f a1 = *(const v4f*)(p + 4);
            v4f a2 = *(const v4f*)(p + 16);
            v4f a3 = *(const v4f*)(p + 20);
            v16h f;
            f[0]=(_Float16)a0.x; f[1]=(_Float16)a0.y; f[2]=(_Float16)a0.z; f[3]=(_Float16)a0.w;
            f[4]=(_Float16)a1.x; f[5]=(_Float16)a1.y; f[6]=(_Float16)a1.z; f[7]=(_Float16)a1.w;
            f[8]=(_Float16)a2.x; f[9]=(_Float16)a2.y; f[10]=(_Float16)a2.z; f[11]=(_Float16)a2.w;
            f[12]=(_Float16)a3.x; f[13]=(_Float16)a3.y; f[14]=(_Float16)a3.z; f[15]=(_Float16)a3.w;
            afrag[t] = f;
        }
        v16h bfrag[2];
        const int bk0 = halfSel * 8;
#pragma unroll
        for (int u = 0; u < 2; ++u) {
            const v4u* q = (const v4u*)&Bs[(wn * 32 + u * 16 + l16) * SSTR + bk0];
            v4u w0 = q[0], w1 = q[1], w2 = q[4], w3 = q[5];
            v8u pk = { sgn_pk(w0.x, w0.y), sgn_pk(w0.z, w0.w),
                       sgn_pk(w1.x, w1.y), sgn_pk(w1.z, w1.w),
                       sgn_pk(w2.x, w2.y), sgn_pk(w2.z, w2.w),
                       sgn_pk(w3.x, w3.y), sgn_pk(w3.z, w3.w) };
            bfrag[u] = __builtin_bit_cast(v16h, pk);
        }
#pragma unroll
        for (int t = 0; t < 4; ++t)
#pragma unroll
            for (int u = 0; u < 2; ++u)
                acc[t][u] = __builtin_amdgcn_wmma_f32_16x16x32_f16(
                    false, afrag[t], false, bfrag[u],
                    (short)0, acc[t][u], false, false);
        __syncthreads();
    }

    const float alpha = wsAlpha[0] * (1.0f / ((float)N_DIM * (float)K_DIM));
#pragma unroll
    for (int u = 0; u < 2; ++u) {
        const int col = nBlock + wn * 32 + u * 16 + l16;
        const float bv = bias[col];
#pragma unroll
        for (int t = 0; t < 4; ++t) {
            const int rowBase = mBlock + wm * 64 + t * 16 + halfSel * 8;
#pragma unroll
            for (int j = 0; j < 8; ++j)
                __builtin_nontemporal_store((acc[t][u][j] + bv) * alpha,
                    &Out[(size_t)(rowBase + j) * N_DIM + col]);
        }
    }
}

// ---------------------------------------------------------------------------
extern "C" void kernel_launch(void* const* d_in, const int* in_sizes, int n_in,
                              void* d_out, int out_size, void* d_ws, size_t ws_size,
                              hipStream_t stream) {
    const float* x    = (const float*)d_in[0];   // [2,4096,4096]
    const float* w    = (const float*)d_in[1];   // [16384,4096]
    const float* bias = (const float*)d_in[2];   // [16384]
    float* out = (float*)d_out;                  // [2,4096,16384]
    float* ws  = (float*)d_ws;

    const size_t xhBytes = (size_t)M_DIM * K_DIM * 2;   //  64 MB
    const size_t whBytes = (size_t)N_DIM * K_DIM * 2;   // 128 MB
    const size_t need    = 1024 + xhBytes + whBytes;

    bitnet_zero_ws<<<1, 64, 0, stream>>>(ws);

    if (ws_size >= need) {
        _Float16* Xh = (_Float16*)((char*)d_ws + 1024);
        _Float16* Wh = (_Float16*)((char*)d_ws + 1024 + xhBytes);

        bitnet_quant_x<<<4096, 256, 0, stream>>>(
            x, Xh, (unsigned int)((size_t)M_DIM * K_DIM / 4));
        bitnet_quant_w<<<4096, 256, 0, stream>>>(
            w, Wh, ws, (unsigned int)((size_t)N_DIM * K_DIM / 4));

        dim3 grid(M_DIM / PBM, N_DIM / PBN);    // M fastest: W panel L2-resident
        bitnet_gemm16<<<grid, 256, 0, stream>>>(Xh, Wh, bias, ws, out);
    } else {
        bitnet_absum<<<2048, 256, 0, stream>>>(
            w, ws, (unsigned int)((size_t)N_DIM * K_DIM / 4));
        dim3 grid(M_DIM / FBM, N_DIM / FBN);
        bitnet_gemm_fb<<<grid, 256, 0, stream>>>(x, w, bias, ws, out);
    }
}
